// MViT_46042049413164
// MI455X (gfx1250) — matrix operand order repaired and checked by reading, not compiled, not run on hardware
//
#include <hip/hip_runtime.h>
#include <hip/hip_bf16.h>
#include <math.h>

// ---------------------------------------------------------------------------
// Types for CDNA5 WMMA (wave32, 16x16x32 bf16 -> f32)
// ---------------------------------------------------------------------------
typedef __attribute__((ext_vector_type(16))) __bf16 v16bf;
typedef __attribute__((ext_vector_type(4)))  __bf16 v4bf;
typedef __attribute__((ext_vector_type(8)))  float  v8f;

union FragU { uint4 u[2]; v16bf v; };

// ---------------------------------------------------------------------------
// WMMA GEMM: C[M,N] = alpha * (A[M,K] @ op(B)) + bias[N]
//   BT==1: B is [N,K]  (C = A @ B^T, the "x @ W^T" form)
//   BT==0: B is [K,N]  (C = A @ B,   the "attn @ V"  form)
// Batched via blockIdx.z with element strides sA/sB/sC.
// Tile: 64x64 per 128-thread block (4 waves). Each wave owns a 32x32 output
// quadrant = 2 A frags x 2 B frags -> 4 WMMAs per 32-K step.
// Interior tiles stage global->LDS via float4 loads + packed bf16 stores;
// edge tiles fall back to a guarded scalar loop.
// ---------------------------------------------------------------------------
#define TM 64
#define TN 64
#define TK 32

template<int BT>
__global__ __launch_bounds__(128) void gemm_wmma(
    const float* __restrict__ A, const float* __restrict__ Bm,
    const float* __restrict__ bias, float* __restrict__ C,
    int M, int N, int K, long sA, long sB, long sC, float alpha)
{
  __shared__ __attribute__((aligned(16))) __bf16 As[TM][TK];
  __shared__ __attribute__((aligned(16))) __bf16 Bs[TN][TK];   // [n][k] always

  const int bz = blockIdx.z;
  A  += (long)bz * sA;
  Bm += (long)bz * sB;
  C  += (long)bz * sC;

  const int m0 = blockIdx.y * TM;
  const int n0 = blockIdx.x * TN;
  const int tid  = threadIdx.x;          // 0..127
  const int wave = tid >> 5;             // 0..3
  const int lane = tid & 31;
  const int l16  = lane & 15;
  const int half = lane >> 4;
  const int wm = (wave >> 1) << 5;       // 0 or 32
  const int wn = (wave & 1) << 5;        // 0 or 32

  const bool fullM = (m0 + TM <= M);
  const bool fullN = (n0 + TN <= N);
  const bool k4    = ((K & 3) == 0);
  const bool n4    = ((N & 3) == 0);

  v8f acc00 = {}, acc01 = {}, acc10 = {}, acc11 = {};

  for (int k0 = 0; k0 < K; k0 += TK) {
    const bool fullK = (k0 + TK <= K);

    // ---------------- stage A (64x32) ----------------
    if (fullM && fullK && k4) {
      #pragma unroll
      for (int r = 0; r < 4; ++r) {
        int idx = (tid + r * 128) * 4;          // 0..2047, step 4
        int m = idx >> 5, k = idx & 31;
        const float4 f = *(const float4*)&A[(long)(m0 + m) * K + (k0 + k)];
        v4bf hv = { (__bf16)f.x, (__bf16)f.y, (__bf16)f.z, (__bf16)f.w };
        *(v4bf*)&As[m][k] = hv;
      }
    } else {
      for (int i = tid; i < TM * TK; i += 128) {
        int m = i >> 5, k = i & 31;
        int gm = m0 + m, gk = k0 + k;
        float v = (gm < M && gk < K) ? A[(long)gm * K + gk] : 0.0f;
        As[m][k] = (__bf16)v;
      }
    }

    // ---------------- stage B (64 n x 32 k, stored [n][k]) ----------------
    if (BT) {
      if (fullN && fullK && k4) {
        #pragma unroll
        for (int r = 0; r < 4; ++r) {
          int idx = (tid + r * 128) * 4;
          int n = idx >> 5, k = idx & 31;
          const float4 f = *(const float4*)&Bm[(long)(n0 + n) * K + (k0 + k)];
          v4bf hv = { (__bf16)f.x, (__bf16)f.y, (__bf16)f.z, (__bf16)f.w };
          *(v4bf*)&Bs[n][k] = hv;
        }
      } else {
        for (int i = tid; i < TN * TK; i += 128) {
          int n = i >> 5, k = i & 31;
          int gn = n0 + n, gk = k0 + k;
          float v = (gn < N && gk < K) ? Bm[(long)gn * K + gk] : 0.0f;
          Bs[n][k] = (__bf16)v;
        }
      }
    } else {
      if (fullN && fullK && n4) {
        // vectorize along n (contiguous in memory), scatter into Bs rows
        #pragma unroll
        for (int r = 0; r < 4; ++r) {
          int g = tid + r * 128;                 // 0..511 float4 groups
          int k = g >> 4, n = (g & 15) * 4;      // 32 k rows x 16 groups
          const float4 f = *(const float4*)&Bm[(long)(k0 + k) * N + (n0 + n)];
          Bs[n + 0][k] = (__bf16)f.x;
          Bs[n + 1][k] = (__bf16)f.y;
          Bs[n + 2][k] = (__bf16)f.z;
          Bs[n + 3][k] = (__bf16)f.w;
        }
      } else {
        for (int i = tid; i < TN * TK; i += 128) {
          int n = i >> 5, k = i & 31;
          int gn = n0 + n, gk = k0 + k;
          float v = (gn < N && gk < K) ? Bm[(long)gk * N + gn] : 0.0f;
          Bs[n][k] = (__bf16)v;
        }
      }
    }
    __syncthreads();

    // ---- fragments per ISA VGPR layouts ----
    FragU a0, a1, b0, b1;
    a0.u[0] = *(const uint4*)&As[wm + l16][8 * half];
    a0.u[1] = *(const uint4*)&As[wm + l16][16 + 8 * half];
    a1.u[0] = *(const uint4*)&As[wm + 16 + l16][8 * half];
    a1.u[1] = *(const uint4*)&As[wm + 16 + l16][16 + 8 * half];
    b0.u[0] = *(const uint4*)&Bs[wn + l16][8 * half];
    b0.u[1] = *(const uint4*)&Bs[wn + l16][16 + 8 * half];
    b1.u[0] = *(const uint4*)&Bs[wn + 16 + l16][8 * half];
    b1.u[1] = *(const uint4*)&Bs[wn + 16 + l16][16 + 8 * half];

    acc00 = __builtin_amdgcn_wmma_f32_16x16x32_bf16(
        false, a0.v, false, b0.v, (short)0, acc00, false, false);
    acc01 = __builtin_amdgcn_wmma_f32_16x16x32_bf16(
        false, a0.v, false, b1.v, (short)0, acc01, false, false);
    acc10 = __builtin_amdgcn_wmma_f32_16x16x32_bf16(
        false, a1.v, false, b0.v, (short)0, acc10, false, false);
    acc11 = __builtin_amdgcn_wmma_f32_16x16x32_bf16(
        false, a1.v, false, b1.v, (short)0, acc11, false, false);

    __syncthreads();
  }

  // ---- store D: lane holds n = wn+l16 (+16); vgpr r holds m = r + 8*half ----
  const int nA = n0 + wn + l16;
  const int nB = nA + 16;
  const float bvA = (bias && nA < N) ? bias[nA] : 0.0f;
  const float bvB = (bias && nB < N) ? bias[nB] : 0.0f;
  #pragma unroll
  for (int r = 0; r < 8; ++r) {
    int m = m0 + wm + r + 8 * half;
    if (m < M) {
      if (nA < N) C[(long)m * N + nA] = alpha * acc00[r] + bvA;
      if (nB < N) C[(long)m * N + nB] = alpha * acc01[r] + bvB;
    }
    int m2 = m + 16;
    if (m2 < M) {
      if (nA < N) C[(long)m2 * N + nA] = alpha * acc10[r] + bvA;
      if (nB < N) C[(long)m2 * N + nB] = alpha * acc11[r] + bvB;
    }
  }
}

// ---------------------------------------------------------------------------
// LayerNorm over last dim C (one thread per token; in-place safe)
// ---------------------------------------------------------------------------
__global__ void ln_kernel(const float* __restrict__ x, const float* __restrict__ w,
                          const float* __restrict__ b, float* __restrict__ y,
                          int Ntok, int C)
{
  int t = blockIdx.x * blockDim.x + threadIdx.x;
  if (t >= Ntok) return;
  const float* xr = x + (long)t * C;
  float* yr = y + (long)t * C;
  float m = 0.f;
  for (int c = 0; c < C; ++c) m += xr[c];
  m /= C;
  float v = 0.f;
  for (int c = 0; c < C; ++c) { float d = xr[c] - m; v += d * d; }
  v /= C;
  float inv = rsqrtf(v + 1e-6f);
  for (int c = 0; c < C; ++c) yr[c] = (xr[c] - m) * inv * w[c] + b[c];
}

// ---------------------------------------------------------------------------
// Patch embed: conv 3->96, kernel (3,7,7), stride (2,4,4), pad (1,3,3)
// writes tokens (B, 1+8*56*56, 96), token 0 = cls
// ---------------------------------------------------------------------------
__global__ void patch_conv_kernel(const float* __restrict__ x,
                                  const float* __restrict__ w,
                                  const float* __restrict__ bias,
                                  const float* __restrict__ cls,
                                  float* __restrict__ tokens, int Bb)
{
  const int C = 96, T = 8, H = 56, W = 56;
  const long NTot = (long)T * H * W + 1;
  long total = (long)Bb * NTot * C;
  long i = (long)blockIdx.x * blockDim.x + threadIdx.x;
  if (i >= total) return;
  int c = i % C;
  long n = (i / C) % NTot;
  int b = (int)(i / (C * NTot));
  if (n == 0) { tokens[i] = cls[c]; return; }
  long g = n - 1;
  int xx = g % W, yy = (g / W) % H, tt = (int)(g / ((long)W * H));
  float s = bias[c];
  for (int ci = 0; ci < 3; ++ci)
    for (int kt = 0; kt < 3; ++kt) {
      int it = tt * 2 - 1 + kt; if (it < 0 || it >= 16) continue;
      for (int ky = 0; ky < 7; ++ky) {
        int iy = yy * 4 - 3 + ky; if (iy < 0 || iy >= 224) continue;
        for (int kx = 0; kx < 7; ++kx) {
          int ix = xx * 4 - 3 + kx; if (ix < 0 || ix >= 224) continue;
          s += x[(((long)b * 3 + ci) * 16 + it) * 224 * 224 + (long)iy * 224 + ix]
             * w[(((long)(c * 3 + ci) * 3 + kt) * 7 + ky) * 7 + kx];
        }
      }
    }
  tokens[i] = s;
}

// ---------------------------------------------------------------------------
// qkv split: (B,N,3*h*hd) -> q/k/v each (B*h, N, hd)
// ---------------------------------------------------------------------------
__global__ void qkv_split_kernel(const float* __restrict__ qkv,
                                 float* __restrict__ q, float* __restrict__ k,
                                 float* __restrict__ v,
                                 int Bb, int N, int h, int hd)
{
  long total = (long)Bb * h * N * hd;
  long i = (long)blockIdx.x * blockDim.x + threadIdx.x;
  if (i >= total) return;
  int c = i % hd;
  long n = (i / hd) % N;
  int hh = (int)((i / ((long)hd * N)) % h);
  int b  = (int)(i / ((long)hd * N * h));
  int Do = h * hd;
  const float* base = qkv + ((long)b * N + n) * 3 * Do;
  long o = (((long)b * h + hh) * N + n) * hd + c;
  q[o] = base[hh * hd + c];
  k[o] = base[Do + hh * hd + c];
  v[o] = base[2 * Do + hh * hd + c];
}

// ---------------------------------------------------------------------------
// Depthwise 3x3x3 pool conv over token grid (cls passthrough), pad 1
// in: (BH, NinTot, hd), out: (BH, NoTot, hd)
// ---------------------------------------------------------------------------
__global__ void pool_conv_kernel(const float* __restrict__ in,
                                 const float* __restrict__ w,
                                 float* __restrict__ out,
                                 int BH, int NinTot, int NoTot, int hd,
                                 int T, int H, int W,
                                 int st, int sh, int sw,
                                 int To, int Ho, int Wo)
{
  long total = (long)BH * NoTot * hd;
  long i = (long)blockIdx.x * blockDim.x + threadIdx.x;
  if (i >= total) return;
  int c = i % hd;
  long n = (i / hd) % NoTot;
  int bh = (int)(i / ((long)hd * NoTot));
  const float* ip = in + (long)bh * NinTot * hd;
  if (n == 0) { out[i] = ip[c]; return; }
  long g = n - 1;
  int x = g % Wo, y = (g / Wo) % Ho, t = (int)(g / ((long)Wo * Ho));
  float s = 0.f;
  for (int dt = 0; dt < 3; ++dt) {
    int it = t * st - 1 + dt; if (it < 0 || it >= T) continue;
    for (int dy = 0; dy < 3; ++dy) {
      int iy = y * sh - 1 + dy; if (iy < 0 || iy >= H) continue;
      for (int dx = 0; dx < 3; ++dx) {
        int ix = x * sw - 1 + dx; if (ix < 0 || ix >= W) continue;
        s += ip[(1 + ((long)it * H + iy) * W + ix) * (long)hd + c]
           * w[c * 27 + dt * 9 + dy * 3 + dx];
      }
    }
  }
  out[i] = s;
}

// ---------------------------------------------------------------------------
// Rel-pos bias: out[bh, g, kj] = dot(qp[bh, 1+g, :], R[relidx(coord,kj), :])
// axis: 0 = t, 1 = h, 2 = w
// ---------------------------------------------------------------------------
__global__ void relbias_kernel(const float* __restrict__ qp,
                               const float* __restrict__ R,
                               float* __restrict__ out,
                               int BH, int NqTot, int hd,
                               int qt, int qh, int qw,
                               int axis, int qn, int kn)
{
  long NqG = (long)qt * qh * qw;
  long total = (long)BH * NqG * kn;
  long i = (long)blockIdx.x * blockDim.x + threadIdx.x;
  if (i >= total) return;
  int kj = i % kn;
  long g = (i / kn) % NqG;
  int bh = (int)(i / ((long)kn * NqG));
  int coord;
  if (axis == 0) coord = (int)(g / ((long)qh * qw));
  else if (axis == 1) coord = (int)((g / qw) % qh);
  else coord = (int)(g % qw);
  float qr = (kn > qn) ? (float)kn / (float)qn : 1.0f;
  float kr = (qn > kn) ? (float)qn / (float)kn : 1.0f;
  int idx = (int)(coord * qr - kj * kr + (kn - 1) * kr + 0.5f);
  const float* qrow = qp + ((long)bh * NqTot + 1 + g) * hd;
  const float* Rrow = R + (long)idx * hd;
  float s = 0.f;
  for (int c = 0; c < hd; ++c) s += qrow[c] * Rrow[c];
  out[i] = s;
}

// ---------------------------------------------------------------------------
// Scatter-add rel biases into attn[.., 1+qi, 1+kj]
// ---------------------------------------------------------------------------
__global__ void add_rel_kernel(float* __restrict__ attn,
                               const float* __restrict__ rt,
                               const float* __restrict__ rh,
                               const float* __restrict__ rw,
                               int BH, int NqTot, int NkTot,
                               int qt, int qh, int qw,
                               int kt, int kh, int kw)
{
  long NqG = (long)qt * qh * qw, NkG = (long)kt * kh * kw;
  long total = (long)BH * NqG * NkG;
  long i = (long)blockIdx.x * blockDim.x + threadIdx.x;
  if (i >= total) return;
  long kj = i % NkG;
  long qi = (i / NkG) % NqG;
  int bh = (int)(i / (NkG * NqG));
  int kx = kj % kw;
  int ky = (kj / kw) % kh;
  int kT = (int)(kj / ((long)kw * kh));
  long qb = (long)bh * NqG + qi;
  float add = rt[qb * kt + kT] + rh[qb * kh + ky] + rw[qb * kw + kx];
  attn[((long)bh * NqTot + 1 + qi) * NkTot + 1 + kj] += add;
}

// ---------------------------------------------------------------------------
// Row softmax (one 256-thread block per row)
// ---------------------------------------------------------------------------
__global__ __launch_bounds__(256) void softmax_kernel(float* __restrict__ attn, int Nk)
{
  long row = blockIdx.x;
  float* a = attn + row * (long)Nk;
  __shared__ float red[256];
  int tid = threadIdx.x;
  float mx = -3.4e38f;
  for (int i = tid; i < Nk; i += 256) mx = fmaxf(mx, a[i]);
  red[tid] = mx; __syncthreads();
  for (int s = 128; s > 0; s >>= 1) { if (tid < s) red[tid] = fmaxf(red[tid], red[tid + s]); __syncthreads(); }
  mx = red[0]; __syncthreads();
  float sum = 0.f;
  for (int i = tid; i < Nk; i += 256) { float e = __expf(a[i] - mx); a[i] = e; sum += e; }
  red[tid] = sum; __syncthreads();
  for (int s = 128; s > 0; s >>= 1) { if (tid < s) red[tid] += red[tid + s]; __syncthreads(); }
  float inv = 1.0f / red[0];
  for (int i = tid; i < Nk; i += 256) a[i] *= inv;
}

// ---------------------------------------------------------------------------
// Elementwise helpers
// ---------------------------------------------------------------------------
__global__ void add_inplace_kernel(float* __restrict__ y, const float* __restrict__ x, long n)
{ long i = (long)blockIdx.x * blockDim.x + threadIdx.x; if (i < n) y[i] += x[i]; }

__global__ void add_out_kernel(float* __restrict__ o, const float* __restrict__ a,
                               const float* __restrict__ b, long n)
{ long i = (long)blockIdx.x * blockDim.x + threadIdx.x; if (i < n) o[i] = a[i] + b[i]; }

__global__ void gelu_kernel(float* __restrict__ x, long n)
{
  long i = (long)blockIdx.x * blockDim.x + threadIdx.x;
  if (i >= n) return;
  float v = x[i];
  x[i] = 0.5f * v * (1.0f + erff(v * 0.70710678118654752f));
}

// (B*h, N, hd) -> (B, N, h*hd)
__global__ void merge_heads_kernel(const float* __restrict__ o, float* __restrict__ o2,
                                   int Bb, int h, int N, int hd)
{
  long total = (long)Bb * h * N * hd;
  long i = (long)blockIdx.x * blockDim.x + threadIdx.x;
  if (i >= total) return;
  int c = i % hd;
  long n = (i / hd) % N;
  int hh = (int)((i / ((long)hd * N)) % h);
  int b  = (int)(i / ((long)hd * N * h));
  o2[((long)b * N + n) * (h * hd) + hh * hd + c] =
      o[(((long)b * h + hh) * N + n) * hd + c];
}

// ---------------------------------------------------------------------------
// MaxPool skip (cls passthrough), window (kt,kh,kw), pad (pt,ph,pw)
// ---------------------------------------------------------------------------
__global__ void maxpool_kernel(const float* __restrict__ in, float* __restrict__ out,
                               int Bb, int C, int T, int H, int W,
                               int st, int sh, int sw,
                               int kt, int kh, int kw,
                               int pt, int ph, int pw,
                               int To, int Ho, int Wo)
{
  long NinTot = (long)T * H * W + 1, NoTot = (long)To * Ho * Wo + 1;
  long total = (long)Bb * NoTot * C;
  long i = (long)blockIdx.x * blockDim.x + threadIdx.x;
  if (i >= total) return;
  int c = i % C;
  long n = (i / C) % NoTot;
  int b = (int)(i / ((long)C * NoTot));
  const float* ip = in + (long)b * NinTot * C;
  if (n == 0) { out[i] = ip[c]; return; }
  long g = n - 1;
  int x = g % Wo, y = (g / Wo) % Ho, t = (int)(g / ((long)Wo * Ho));
  float m = -3.4e38f;
  for (int dt = 0; dt < kt; ++dt) {
    int it = t * st - pt + dt; if (it < 0 || it >= T) continue;
    for (int dy = 0; dy < kh; ++dy) {
      int iy = y * sh - ph + dy; if (iy < 0 || iy >= H) continue;
      for (int dx = 0; dx < kw; ++dx) {
        int ix = x * sw - pw + dx; if (ix < 0 || ix >= W) continue;
        m = fmaxf(m, ip[(1 + ((long)it * H + iy) * W + ix) * (long)C + c]);
      }
    }
  }
  out[i] = m;
}

// ---------------------------------------------------------------------------
// Host orchestration
// ---------------------------------------------------------------------------
struct BP {
  const float *bproj_b, *bproj_w, *fc1_b, *fc1_w, *fc2_b, *fc2_w;
  const float *n1_b, *n1_w, *n2_b, *n2_w, *nk_b, *nk_w, *nq_b, *nq_w, *nv_b, *nv_w;
  const float *pk_w, *pq_w, *proj_b, *proj_w, *pv_w, *qkv_b, *qkv_w;
  const float *rel_h, *rel_t, *rel_w;
};

struct BCfg { int d, dout, heads, has_sq; int sq[3]; int skv[3]; };

extern "C" void kernel_launch(void* const* d_in, const int* in_sizes, int n_in,
                              void* d_out, int out_size, void* d_ws, size_t ws_size,
                              hipStream_t stream)
{
  (void)in_sizes; (void)n_in; (void)out_size; (void)ws_size;

  // --- input walking: jax pytree order (dict keys alphabetical) ---
  int ii = 0;
  auto NXT = [&]() { return (const float*)d_in[ii++]; };

  BP bps[4];
  for (int bl = 0; bl < 4; ++bl) {
    bool ex = (bl == 1 || bl == 3);   // has bproj_* , nq_* , pq_w
    BP& p = bps[bl];
    p.bproj_b = p.bproj_w = p.nq_b = p.nq_w = p.pq_w = nullptr;
    if (ex) { p.bproj_b = NXT(); p.bproj_w = NXT(); }
    p.fc1_b = NXT(); p.fc1_w = NXT(); p.fc2_b = NXT(); p.fc2_w = NXT();
    p.n1_b = NXT(); p.n1_w = NXT(); p.n2_b = NXT(); p.n2_w = NXT();
    p.nk_b = NXT(); p.nk_w = NXT();
    if (ex) { p.nq_b = NXT(); p.nq_w = NXT(); }
    p.nv_b = NXT(); p.nv_w = NXT();
    p.pk_w = NXT();
    if (ex) { p.pq_w = NXT(); }
    p.proj_b = NXT(); p.proj_w = NXT(); p.pv_w = NXT();
    p.qkv_b = NXT(); p.qkv_w = NXT();
    p.rel_h = NXT(); p.rel_t = NXT(); p.rel_w = NXT();
  }
  const float* cls     = NXT();
  const float* norm_b  = NXT();
  const float* norm_w  = NXT();
  const float* patch_b = NXT();
  const float* patch_w = NXT();
  const float* xin     = NXT();

  // --- workspace bump allocator (float counts) ---
  size_t off = 0;
  auto alloc = [&](size_t nfloats) -> float* {
    float* p = (float*)((char*)d_ws + off);
    off += ((nfloats * sizeof(float)) + 255) & ~(size_t)255;
    return p;
  };
  float* Xa   = alloc(4900000);
  float* Xb   = alloc(4900000);
  float* xn   = alloc(4900000);
  float* qkvb = alloc(29000000);
  float* qb   = alloc(9700000);
  float* kb   = alloc(9700000);
  float* vb   = alloc(9700000);
  float* qpb  = alloc(2600000);
  float* kpb  = alloc(1300000);
  float* vpb  = alloc(1300000);
  float* attn = alloc(39500000);
  float* ob   = alloc(4900000);
  float* o2b  = alloc(4900000);
  float* xsb  = alloc(9700000);
  float* xpb  = alloc(2600000);
  float* mlpb = alloc(19400000);
  float* rtb  = alloc(600000);
  float* rhb  = alloc(600000);
  float* rwb  = alloc(600000);

  auto nb = [](long n) { return (unsigned)((n + 255) / 256); };

  auto gemm = [&](const float* A, const float* Bm, const float* bias, float* C,
                  int M, int N, int K, long sA, long sB, long sC, int batch,
                  float alpha, bool BT) {
    dim3 g((N + TN - 1) / TN, (M + TM - 1) / TM, batch);
    if (BT) gemm_wmma<1><<<g, 128, 0, stream>>>(A, Bm, bias, C, M, N, K, sA, sB, sC, alpha);
    else    gemm_wmma<0><<<g, 128, 0, stream>>>(A, Bm, bias, C, M, N, K, sA, sB, sC, alpha);
  };

  const BCfg cfgs[4] = {
    { 96,  96, 1, 0, {1, 1, 1}, {1, 8, 8} },
    { 96, 192, 2, 1, {1, 2, 2}, {1, 4, 4} },
    {192, 192, 2, 0, {1, 1, 1}, {1, 4, 4} },
    {192, 384, 4, 1, {1, 2, 2}, {1, 2, 2} },
  };

  const int Bq = 2;

  // --- patch embed -> tokens in Xa ---
  {
    long total = (long)Bq * (8 * 56 * 56 + 1) * 96;
    patch_conv_kernel<<<nb(total), 256, 0, stream>>>(xin, patch_w, patch_b, cls, Xa, Bq);
  }

  float* X  = Xa;
  float* Xn = Xb;
  int thw[3] = {8, 56, 56};

  for (int bl = 0; bl < 4; ++bl) {
    const BCfg& c = cfgs[bl];
    const BP&   p = bps[bl];
    const int d = c.d, Do = c.dout, h = c.heads, hd = Do / h;
    const int T = thw[0], H = thw[1], W = thw[2];
    const int Nin = T * H * W + 1;
    int qthw[3], kthw[3];
    for (int j = 0; j < 3; ++j) {
      qthw[j] = c.has_sq ? ((thw[j] - 1) / c.sq[j] + 1) : thw[j];
      kthw[j] = (thw[j] - 1) / c.skv[j] + 1;
    }
    const int NqG = qthw[0] * qthw[1] * qthw[2], Nq = NqG + 1;
    const int NkG = kthw[0] * kthw[1] * kthw[2], Nk = NkG + 1;
    const int BH = Bq * h;

    // 1. LN1
    ln_kernel<<<nb(Bq * Nin), 256, 0, stream>>>(X, p.n1_w, p.n1_b, xn, Bq * Nin, d);
    // 2. qkv = xn @ qkv_w^T + b
    gemm(xn, p.qkv_w, p.qkv_b, qkvb, Bq * Nin, 3 * Do, d, 0, 0, 0, 1, 1.0f, true);
    // 3. split heads
    qkv_split_kernel<<<nb((long)BH * Nin * hd), 256, 0, stream>>>(qkvb, qb, kb, vb, Bq, Nin, h, hd);

    // 4. pooling convs + LN
    float* qp;
    if (c.has_sq) {
      pool_conv_kernel<<<nb((long)BH * Nq * hd), 256, 0, stream>>>(
          qb, p.pq_w, qpb, BH, Nin, Nq, hd, T, H, W,
          c.sq[0], c.sq[1], c.sq[2], qthw[0], qthw[1], qthw[2]);
      ln_kernel<<<nb(BH * Nq), 256, 0, stream>>>(qpb, p.nq_w, p.nq_b, qpb, BH * Nq, hd);
      qp = qpb;
    } else {
      qp = qb;
    }
    pool_conv_kernel<<<nb((long)BH * Nk * hd), 256, 0, stream>>>(
        kb, p.pk_w, kpb, BH, Nin, Nk, hd, T, H, W,
        c.skv[0], c.skv[1], c.skv[2], kthw[0], kthw[1], kthw[2]);
    ln_kernel<<<nb(BH * Nk), 256, 0, stream>>>(kpb, p.nk_w, p.nk_b, kpb, BH * Nk, hd);
    pool_conv_kernel<<<nb((long)BH * Nk * hd), 256, 0, stream>>>(
        vb, p.pv_w, vpb, BH, Nin, Nk, hd, T, H, W,
        c.skv[0], c.skv[1], c.skv[2], kthw[0], kthw[1], kthw[2]);
    ln_kernel<<<nb(BH * Nk), 256, 0, stream>>>(vpb, p.nv_w, p.nv_b, vpb, BH * Nk, hd);

    // 5. attention scores: attn = (q*scale) @ k^T   (batched over BH)
    const float alpha = 1.0f / sqrtf((float)hd);
    gemm(qp, kpb, nullptr, attn, Nq, Nk, hd,
         (long)Nq * hd, (long)Nk * hd, (long)Nq * Nk, BH, alpha, true);

    // 6. relative position biases
    relbias_kernel<<<nb((long)BH * NqG * kthw[0]), 256, 0, stream>>>(
        qp, p.rel_t, rtb, BH, Nq, hd, qthw[0], qthw[1], qthw[2], 0, qthw[0], kthw[0]);
    relbias_kernel<<<nb((long)BH * NqG * kthw[1]), 256, 0, stream>>>(
        qp, p.rel_h, rhb, BH, Nq, hd, qthw[0], qthw[1], qthw[2], 1, qthw[1], kthw[1]);
    relbias_kernel<<<nb((long)BH * NqG * kthw[2]), 256, 0, stream>>>(
        qp, p.rel_w, rwb, BH, Nq, hd, qthw[0], qthw[1], qthw[2], 2, qthw[2], kthw[2]);
    add_rel_kernel<<<nb((long)BH * NqG * NkG), 256, 0, stream>>>(
        attn, rtb, rhb, rwb, BH, Nq, Nk,
        qthw[0], qthw[1], qthw[2], kthw[0], kthw[1], kthw[2]);

    // 7. softmax over keys
    softmax_kernel<<<(unsigned)(BH * Nq), 256, 0, stream>>>(attn, Nk);

    // 8. o = attn @ v   (B is [K,N])
    gemm(attn, vpb, nullptr, ob, Nq, hd, Nk,
         (long)Nq * Nk, (long)Nk * hd, (long)Nq * hd, BH, 1.0f, false);
    // 9. o += pooled q
    add_inplace_kernel<<<nb((long)BH * Nq * hd), 256, 0, stream>>>(ob, qp, (long)BH * Nq * hd);
    // 10. merge heads -> (B, Nq, Do)
    merge_heads_kernel<<<nb((long)BH * Nq * hd), 256, 0, stream>>>(ob, o2b, Bq, h, Nq, hd);
    // 11. o = o2 @ proj^T + b  (reuse ob)
    gemm(o2b, p.proj_w, p.proj_b, ob, Bq * Nq, Do, Do, 0, 0, 0, 1, 1.0f, true);

    // 12. skip path
    const float* skip = X;
    if (d != Do) {
      gemm(xn, p.bproj_w, p.bproj_b, xsb, Bq * Nin, Do, d, 0, 0, 0, 1, 1.0f, true);
      skip = xsb;
    }
    if (c.has_sq) {
      int kd[3], pd[3];
      for (int j = 0; j < 3; ++j) { kd[j] = c.sq[j] > 1 ? c.sq[j] + 1 : c.sq[j]; pd[j] = kd[j] / 2; }
      maxpool_kernel<<<nb((long)Bq * Nq * Do), 256, 0, stream>>>(
          skip, xpb, Bq, Do, T, H, W,
          c.sq[0], c.sq[1], c.sq[2], kd[0], kd[1], kd[2], pd[0], pd[1], pd[2],
          qthw[0], qthw[1], qthw[2]);
      skip = xpb;
    }
    // 13. Xn = skip + o
    add_out_kernel<<<nb((long)Bq * Nq * Do), 256, 0, stream>>>(Xn, skip, ob, (long)Bq * Nq * Do);
    // 14. LN2
    ln_kernel<<<nb(Bq * Nq), 256, 0, stream>>>(Xn, p.n2_w, p.n2_b, xn, Bq * Nq, Do);
    // 15. MLP fc1 + exact GELU
    gemm(xn, p.fc1_w, p.fc1_b, mlpb, Bq * Nq, 4 * Do, Do, 0, 0, 0, 1, 1.0f, true);
    gelu_kernel<<<nb((long)Bq * Nq * 4 * Do), 256, 0, stream>>>(mlpb, (long)Bq * Nq * 4 * Do);
    // 16. fc2
    gemm(mlpb, p.fc2_w, p.fc2_b, o2b, Bq * Nq, Do, 4 * Do, 0, 0, 0, 1, 1.0f, true);
    // 17. residual
    add_inplace_kernel<<<nb((long)Bq * Nq * Do), 256, 0, stream>>>(Xn, o2b, (long)Bq * Nq * Do);

    float* tmp = X; X = Xn; Xn = tmp;
    thw[0] = qthw[0]; thw[1] = qthw[1]; thw[2] = qthw[2];
  }

  // final layernorm -> d_out  (2, 1569, 384)
  ln_kernel<<<nb(Bq * 1569), 256, 0, stream>>>(X, norm_w, norm_b, (float*)d_out, Bq * 1569, 384);
}
